// MultiHeadAttention_30846455119878
// MI455X (gfx1250) — matrix operand order repaired and checked by reading, not compiled, not run on hardware
//
#include <hip/hip_runtime.h>

// ---------------------------------------------------------------------------
// MultiHeadAttention for MI455X (gfx1250), wave32 + v_wmma_f32_16x16x32_bf16,
// with GLOBAL_LOAD_ASYNC_TO_LDS for direct-copy tiles.
// B=4, T=2048, C=1024, H=16, D=64.
// ---------------------------------------------------------------------------

typedef __attribute__((ext_vector_type(16))) __bf16 v16bf;
typedef __attribute__((ext_vector_type(8)))  float  v8f;
typedef __attribute__((ext_vector_type(4)))  unsigned int v4ui;
typedef __attribute__((ext_vector_type(4)))  int v4i;

#if defined(__has_builtin)
#  if __has_builtin(__builtin_amdgcn_global_load_async_to_lds_b128)
#    define HAVE_ASYNC_LDS 1
#  endif
#endif

union FragAB { v16bf v; unsigned int u[8]; };
union FragC  { v8f   v; float f[8]; };

__device__ __forceinline__ unsigned short f32_to_bf16(float f) {
  unsigned int u = __builtin_bit_cast(unsigned int, f);
  u += 0x7fffu + ((u >> 16) & 1u);           // round-to-nearest-even
  return (unsigned short)(u >> 16);
}

// A-matrix (16x32 bf16) uint index inside a 16-uint row for VGPR v, half hl.
__device__ __forceinline__ int a_uidx(int v, int hl) {
  return v + 8 * hl;
}

__device__ __forceinline__ v8f wmma_bf16(const FragAB& a, const FragAB& b, v8f c) {
  return __builtin_amdgcn_wmma_f32_16x16x32_bf16(false, a.v, false, b.v,
                                                 (short)0, c, false, false);
}

// 16-byte global -> LDS copy: async DMA path on gfx1250 when the builtin
// exists, otherwise a b128 register round-trip.
__device__ __forceinline__ void copy16_g2lds(unsigned int* lds_dst,
                                             const unsigned int* gsrc) {
#ifdef HAVE_ASYNC_LDS
  __builtin_amdgcn_global_load_async_to_lds_b128(
      (__attribute__((address_space(1))) v4i*)(gsrc),
      (__attribute__((address_space(3))) v4i*)(lds_dst), 0, 0);
#else
  *(v4ui*)lds_dst = *(const v4ui*)gsrc;
#endif
}

__device__ __forceinline__ void async_fence() {
#ifdef HAVE_ASYNC_LDS
#  if __has_builtin(__builtin_amdgcn_s_wait_asynccnt)
  __builtin_amdgcn_s_wait_asynccnt(0);
#  else
  asm volatile("s_wait_asynccnt 0x0" ::: "memory");
#  endif
#endif
}

// ---------------------------------------------------------------------------
// Kernel 1: f32 -> bf16 bit conversion (grid-stride)
// ---------------------------------------------------------------------------
__global__ void cvt_f32_bf16(const float* __restrict__ src,
                             unsigned short* __restrict__ dst, int n) {
  int i = blockIdx.x * blockDim.x + threadIdx.x;
  int stride = gridDim.x * blockDim.x;
  for (; i < n; i += stride) dst[i] = f32_to_bf16(src[i]);
}

// ---------------------------------------------------------------------------
// Kernel 2: QKV projection.  out[t,d] = sum_c x[t,c] * W[h,c,d]
//   A = xb [8192,1024] bf16, B = W[h] [1024,64] bf16, out bf16 [B,H,T,D].
//   grid.x = 32 (M blocks of 256), grid.y = 48 (16 heads x {Q,K,V}).
//   Each wave: 32 rows (2 A-frags) x 64 cols (4 B-frags) = 8 WMMA / k-step.
// ---------------------------------------------------------------------------
__global__ __launch_bounds__(256) void qkv_gemm(
    const unsigned short* __restrict__ xb,
    const unsigned short* __restrict__ Wq,
    const unsigned short* __restrict__ Wk,
    const unsigned short* __restrict__ Wv,
    unsigned short* __restrict__ Qo,
    unsigned short* __restrict__ Ko,
    unsigned short* __restrict__ Vo) {
  __shared__ __align__(16) unsigned int sA[256 * 16];  // 256 rows x 32 bf16
  __shared__ __align__(16) unsigned int sB[64 * 16];   // transposed 64n x 32k

  const int tid = threadIdx.x;
  const int wave = tid >> 5;
  const int lane = tid & 31;
  const int hl = lane >> 4;
  const int ln = lane & 15;

  const int m0 = blockIdx.x * 256;
  const int h = blockIdx.y & 15;
  const int sel = blockIdx.y >> 4;  // 0=Q 1=K 2=V

  const unsigned short* W = (sel == 0) ? Wq : (sel == 1) ? Wk : Wv;
  unsigned short* Out = (sel == 0) ? Qo : (sel == 1) ? Ko : Vo;
  const unsigned short* Wh = W + (size_t)h * (1024 * 64);
  const unsigned int* x32 = (const unsigned int*)xb;

  FragC acc[2][4];
#pragma unroll
  for (int mt = 0; mt < 2; ++mt)
#pragma unroll
    for (int i = 0; i < 4; ++i)
#pragma unroll
      for (int r = 0; r < 8; ++r) acc[mt][i].f[r] = 0.0f;

  for (int k0 = 0; k0 < 1024; k0 += 32) {
    __syncthreads();
#pragma unroll
    for (int i = 0; i < 4; ++i) {  // A tile: 1024 x 16B chunks / 256 thr
      int idx = tid + i * 256;
      int row = idx >> 2, c = idx & 3;
      copy16_g2lds(&sA[idx * 4],
                   &x32[(size_t)(m0 + row) * 512 + (k0 >> 1) + c * 4]);
    }
#pragma unroll
    for (int i = 0; i < 4; ++i) {  // B tile transposed: 1024 uints
      int idx = tid + i * 256;
      int n = idx >> 4, kk = idx & 15;
      unsigned int lo = Wh[(size_t)(k0 + 2 * kk) * 64 + n];
      unsigned int hi = Wh[(size_t)(k0 + 2 * kk + 1) * 64 + n];
      sB[idx] = lo | (hi << 16);
    }
    if (k0 + 32 < 1024)  // gfx1250 global_prefetch of next W tile
      __builtin_prefetch(&Wh[(size_t)(k0 + 32 + (tid >> 3)) * 64 + (tid & 7) * 8],
                         0, 1);
    async_fence();
    __syncthreads();

    FragAB a0, a1;
#pragma unroll
    for (int v = 0; v < 8; ++v) {
      int ui = a_uidx(v, hl);
      a0.u[v] = sA[(wave * 32 + ln) * 16 + ui];
      a1.u[v] = sA[(wave * 32 + 16 + ln) * 16 + ui];
    }
#pragma unroll
    for (int nt = 0; nt < 4; ++nt) {
      FragAB b;
#pragma unroll
      for (int v = 0; v < 8; ++v)
        b.u[v] = sB[(nt * 16 + ln) * 16 + v + 8 * hl];
      acc[0][nt].v = wmma_bf16(a0, b, acc[0][nt].v);
      acc[1][nt].v = wmma_bf16(a1, b, acc[1][nt].v);
    }
  }

#pragma unroll
  for (int mt = 0; mt < 2; ++mt)
#pragma unroll
    for (int nt = 0; nt < 4; ++nt)
#pragma unroll
      for (int r = 0; r < 8; ++r) {
        int m = wave * 32 + mt * 16 + r + 8 * hl;
        int row = m0 + m;
        int bb = row >> 11, t = row & 2047;
        int d = nt * 16 + ln;
        Out[((size_t)((bb * 16 + h) * 2048 + t)) * 64 + d] =
            f32_to_bf16(acc[mt][nt].f[r]);
      }
}

// ---------------------------------------------------------------------------
// Kernel 3: causal flash attention per (b,h,128-row Q block).
//   Q,K,V bf16 [B*H, T, 64]; att bf16 [B, T, H*64] (head-major concat).
//   grid.x = B*H*(T/128) = 1024. 8 waves: each owns 16 query rows.
// ---------------------------------------------------------------------------
__global__ __launch_bounds__(256) void flash_attn(
    const unsigned short* __restrict__ Q,
    const unsigned short* __restrict__ K,
    const unsigned short* __restrict__ V,
    unsigned short* __restrict__ att) {
  __shared__ __align__(16) unsigned int sK[32 * 32];  // K tile [32 s][64 d]
  __shared__ __align__(16) unsigned int sV[64 * 16];  // V^T [64 d][32 s]
  __shared__ unsigned short sP[8][16][32];            // per-wave P scratch

  const int tid = threadIdx.x;
  const int wave = tid >> 5;
  const int lane = tid & 31;
  const int hl = lane >> 4;
  const int ln = lane & 15;

  const int qblk = blockIdx.x & 15;
  const int h = (blockIdx.x >> 4) & 15;
  const int bb = blockIdx.x >> 8;
  const int bh = bb * 16 + h;
  const int q0 = qblk * 128;

  const unsigned int* Q32 = (const unsigned int*)Q + (size_t)bh * 2048 * 32;
  const unsigned int* K32 = (const unsigned int*)K + (size_t)bh * 2048 * 32;
  const unsigned short* Vh = V + (size_t)bh * 2048 * 64;

  // Q fragments (d 0..31 and 32..63) live in registers for the whole pass.
  const int qrowA = q0 + wave * 16 + ln;  // A-layout: lane -> row (both halves)
  FragAB qa0, qa1;
#pragma unroll
  for (int v = 0; v < 8; ++v) {
    int ui = a_uidx(v, hl);
    qa0.u[v] = Q32[(size_t)qrowA * 32 + ui];
    qa1.u[v] = Q32[(size_t)qrowA * 32 + 16 + ui];
  }

  float mi[8], li[8];
  FragC acc[4];
#pragma unroll
  for (int r = 0; r < 8; ++r) { mi[r] = -__builtin_inff(); li[r] = 0.0f; }
#pragma unroll
  for (int i = 0; i < 4; ++i)
#pragma unroll
    for (int r = 0; r < 8; ++r) acc[i].f[r] = 0.0f;

  const float scale = 1.0f / 32.0f;  // 1/sqrt(C) = 1/sqrt(1024)
  const int qmax_wave = q0 + wave * 16 + 15;
  const int send = q0 + 128;

  for (int s0 = 0; s0 < send; s0 += 32) {
    __syncthreads();
    {  // K tile: 256 x 16B chunks, one per thread (async copy path)
      int row = tid >> 3, c = tid & 7;
      copy16_g2lds(&sK[row * 32 + c * 4],
                   &K32[(size_t)(s0 + row) * 32 + c * 4]);
    }
#pragma unroll
    for (int i = 0; i < 4; ++i) {  // V tile transposed: 1024 uints
      int idx = tid + i * 256;
      int d = idx >> 4, kk = idx & 15;
      unsigned int lo = Vh[(size_t)(s0 + 2 * kk) * 64 + d];
      unsigned int hi = Vh[(size_t)(s0 + 2 * kk + 1) * 64 + d];
      sV[idx] = lo | (hi << 16);
    }
    if (s0 + 32 < send)
      __builtin_prefetch(&K32[(size_t)(s0 + 32 + (tid >> 3)) * 32 + (tid & 7) * 4],
                         0, 1);
    async_fence();
    __syncthreads();

    if (s0 <= qmax_wave) {
      // S = Q * K^T  (two 16x16 score tiles, contraction over d=64)
      FragC S[2];
#pragma unroll
      for (int nt = 0; nt < 2; ++nt) {
#pragma unroll
        for (int r = 0; r < 8; ++r) S[nt].f[r] = 0.0f;
        FragAB kb0, kb1;
#pragma unroll
        for (int v = 0; v < 8; ++v) {
          kb0.u[v] = sK[(nt * 16 + ln) * 32 + v + 8 * hl];
          kb1.u[v] = sK[(nt * 16 + ln) * 32 + 16 + v + 8 * hl];
        }
        S[nt].v = wmma_bf16(qa0, kb0, S[nt].v);
        S[nt].v = wmma_bf16(qa1, kb1, S[nt].v);
      }

      // Causal mask + online softmax (rows m = r + 8*hl spread over 16 lanes)
#pragma unroll
      for (int r = 0; r < 8; ++r) {
        int m = r + 8 * hl;
        int qi = q0 + wave * 16 + m;
        float sa = S[0].f[r] * scale;
        float sb = S[1].f[r] * scale;
        if (s0 + ln > qi)      sa = -__builtin_inff();
        if (s0 + 16 + ln > qi) sb = -__builtin_inff();
        float rm = fmaxf(sa, sb);
        rm = fmaxf(rm, __shfl_xor(rm, 1));
        rm = fmaxf(rm, __shfl_xor(rm, 2));
        rm = fmaxf(rm, __shfl_xor(rm, 4));
        rm = fmaxf(rm, __shfl_xor(rm, 8));
        float mn = fmaxf(mi[r], rm);
        float alpha = __expf(mi[r] - mn);
        float ea = __expf(sa - mn);
        float eb = __expf(sb - mn);
        float rs = ea + eb;
        rs += __shfl_xor(rs, 1);
        rs += __shfl_xor(rs, 2);
        rs += __shfl_xor(rs, 4);
        rs += __shfl_xor(rs, 8);
        li[r] = li[r] * alpha + rs;
        mi[r] = mn;
        acc[0].f[r] *= alpha;
        acc[1].f[r] *= alpha;
        acc[2].f[r] *= alpha;
        acc[3].f[r] *= alpha;
        sP[wave][m][ln] = f32_to_bf16(ea);
        sP[wave][m][16 + ln] = f32_to_bf16(eb);
      }

      // Re-load P in A-layout, then O += P * V (contraction over s=32)
      FragAB pfrag;
      const unsigned int* sP32 = (const unsigned int*)&sP[wave][0][0];
#pragma unroll
      for (int v = 0; v < 8; ++v)
        pfrag.u[v] = sP32[ln * 16 + a_uidx(v, hl)];
#pragma unroll
      for (int nt = 0; nt < 4; ++nt) {
        FragAB vb;
#pragma unroll
        for (int v = 0; v < 8; ++v)
          vb.u[v] = sV[(nt * 16 + ln) * 16 + v + 8 * hl];
        acc[nt].v = wmma_bf16(pfrag, vb, acc[nt].v);
      }
    }
  }

  // Normalize and store head-major concat att[b, t, h*64 + d]
#pragma unroll
  for (int nt = 0; nt < 4; ++nt)
#pragma unroll
    for (int r = 0; r < 8; ++r) {
      int m = r + 8 * hl;
      int t = q0 + wave * 16 + m;
      float o = acc[nt].f[r] / li[r];
      att[((size_t)(bb * 2048 + t)) * 1024 + h * 64 + nt * 16 + ln] =
          f32_to_bf16(o);
    }
}

// ---------------------------------------------------------------------------
// Kernel 4: output projection.  out = att[8192,1024] @ Wo[1024,1024] + bo
//   grid (32 M-blocks of 256, 16 N-blocks of 64). f32 output.
//   Each wave: 32 rows x 64 cols = 8 WMMA / k-step.
// ---------------------------------------------------------------------------
__global__ __launch_bounds__(256) void out_proj(
    const unsigned short* __restrict__ A,
    const unsigned short* __restrict__ Wo,
    const float* __restrict__ bo,
    float* __restrict__ out) {
  __shared__ __align__(16) unsigned int sA[256 * 16];
  __shared__ __align__(16) unsigned int sB[64 * 16];  // transposed Wo tile

  const int tid = threadIdx.x;
  const int wave = tid >> 5;
  const int lane = tid & 31;
  const int hl = lane >> 4;
  const int ln = lane & 15;

  const int m0 = blockIdx.x * 256;
  const int n0 = blockIdx.y * 64;
  const unsigned int* A32 = (const unsigned int*)A;

  FragC acc[2][4];
#pragma unroll
  for (int mt = 0; mt < 2; ++mt)
#pragma unroll
    for (int i = 0; i < 4; ++i)
#pragma unroll
      for (int r = 0; r < 8; ++r) acc[mt][i].f[r] = 0.0f;

  for (int k0 = 0; k0 < 1024; k0 += 32) {
    __syncthreads();
#pragma unroll
    for (int i = 0; i < 4; ++i) {
      int idx = tid + i * 256;
      int row = idx >> 2, c = idx & 3;
      copy16_g2lds(&sA[idx * 4],
                   &A32[(size_t)(m0 + row) * 512 + (k0 >> 1) + c * 4]);
    }
#pragma unroll
    for (int i = 0; i < 4; ++i) {
      int idx = tid + i * 256;
      int n = idx >> 4, kk = idx & 15;
      unsigned int lo = Wo[(size_t)(k0 + 2 * kk) * 1024 + n0 + n];
      unsigned int hi = Wo[(size_t)(k0 + 2 * kk + 1) * 1024 + n0 + n];
      sB[idx] = lo | (hi << 16);
    }
    if (k0 + 32 < 1024)
      __builtin_prefetch(&Wo[(size_t)(k0 + 32 + (tid >> 3)) * 1024 + n0 +
                             (tid & 7) * 8], 0, 1);
    async_fence();
    __syncthreads();

    FragAB a0, a1;
#pragma unroll
    for (int v = 0; v < 8; ++v) {
      int ui = a_uidx(v, hl);
      a0.u[v] = sA[(wave * 32 + ln) * 16 + ui];
      a1.u[v] = sA[(wave * 32 + 16 + ln) * 16 + ui];
    }
#pragma unroll
    for (int nt = 0; nt < 4; ++nt) {
      FragAB b;
#pragma unroll
      for (int v = 0; v < 8; ++v)
        b.u[v] = sB[(nt * 16 + ln) * 16 + v + 8 * hl];
      acc[0][nt].v = wmma_bf16(a0, b, acc[0][nt].v);
      acc[1][nt].v = wmma_bf16(a1, b, acc[1][nt].v);
    }
  }

#pragma unroll
  for (int mt = 0; mt < 2; ++mt)
#pragma unroll
    for (int nt = 0; nt < 4; ++nt)
#pragma unroll
      for (int r = 0; r < 8; ++r) {
        int row = m0 + wave * 32 + mt * 16 + r + 8 * hl;
        int n = n0 + nt * 16 + ln;
        out[(size_t)row * 1024 + n] = acc[mt][nt].f[r] + bo[n];
      }
}

// ---------------------------------------------------------------------------
// Launch
// ---------------------------------------------------------------------------
extern "C" void kernel_launch(void* const* d_in, const int* in_sizes, int n_in,
                              void* d_out, int out_size, void* d_ws,
                              size_t ws_size, hipStream_t stream) {
  const float* x  = (const float*)d_in[0];
  const float* Wq = (const float*)d_in[1];
  const float* Wk = (const float*)d_in[2];
  const float* Wv = (const float*)d_in[3];
  const float* Wo = (const float*)d_in[4];
  const float* bo = (const float*)d_in[5];
  float* out = (float*)d_out;

  constexpr size_t N_X = (size_t)8192 * 1024;     // x / att elements
  constexpr size_t N_W = (size_t)16 * 1024 * 64;  // per-head weight stack
  constexpr size_t N_WO = (size_t)1024 * 1024;
  constexpr size_t N_QKV = (size_t)4 * 16 * 2048 * 64;

  unsigned short* p = (unsigned short*)d_ws;
  unsigned short* xb  = p;            p += N_X;
  unsigned short* Wqb = p;            p += N_W;
  unsigned short* Wkb = p;            p += N_W;
  unsigned short* Wvb = p;            p += N_W;
  unsigned short* Wob = p;            p += N_WO;
  unsigned short* Qb  = p;            p += N_QKV;
  unsigned short* Kb  = p;            p += N_QKV;
  unsigned short* Vb  = p;            p += N_QKV;
  unsigned short* attb = p;           p += N_X;

  cvt_f32_bf16<<<2048, 256, 0, stream>>>(x,  xb,  (int)N_X);
  cvt_f32_bf16<<<512,  256, 0, stream>>>(Wq, Wqb, (int)N_W);
  cvt_f32_bf16<<<512,  256, 0, stream>>>(Wk, Wkb, (int)N_W);
  cvt_f32_bf16<<<512,  256, 0, stream>>>(Wv, Wvb, (int)N_W);
  cvt_f32_bf16<<<512,  256, 0, stream>>>(Wo, Wob, (int)N_WO);

  qkv_gemm<<<dim3(32, 48), 256, 0, stream>>>(xb, Wqb, Wkb, Wvb, Qb, Kb, Vb);
  flash_attn<<<1024, 256, 0, stream>>>(Qb, Kb, Vb, attb);
  out_proj<<<dim3(32, 16), 256, 0, stream>>>(attb, Wob, bo, out);
}
